// TemplateEmbedder_20873541058564
// MI455X (gfx1250) — matrix operand order repaired and checked by reading, not compiled, not run on hardware
//
#include <hip/hip_runtime.h>
#include <hip/hip_bf16.h>
#include <math.h>

// ---------------------------------------------------------------------------
// TemplateEmbedder for MI455X (gfx1250), wave32 + WMMA f32_16x16x32_f16.
// All GEMM-shaped work runs on v_wmma_f32_16x16x32_f16 with f32 accumulate.
// GEMMs are fully templated so epilogues/gating are branch-free and all
// load offsets are compile-time immediates (deep MLP of outstanding loads).
// ---------------------------------------------------------------------------

typedef __attribute__((ext_vector_type(16))) _Float16 v16h;
typedef __attribute__((ext_vector_type(8)))  float    v8f;

#define NSEQ 160
#define NN   25600          // 160*160
#define EPSL 1e-5f

__device__ __forceinline__ float4 ld4(const float* p) {
  return *(const float4*)p;
}

// ---------------- generic fused GEMM: D = epi( f16(A[M x K]) @ f16(W[K x N]) )
// lda == K, ldw == ldd == ldg == NOUT (true for every call site).
// EPI: 0 store, 1 sigmoid-store, 2 dst += acc*gate, 3 dst = acc*dst, 4 dst += acc
// AOP: 0 none, 1 relu(0.25*x) on A.  HASAG: multiply A elementwise by agate.
template <int K, int NOUT, int EPI, bool HASAG, int AOP>
__global__ __launch_bounds__(128) void k_gemm(
    const float* __restrict__ A, const float* __restrict__ W,
    float* __restrict__ D, const float* __restrict__ gate,
    const float* __restrict__ agate)
{
  const int wave = threadIdx.x >> 5;
  const int lane = threadIdx.x & 31;
  const int row0 = blockIdx.x * 16;
  const int col0 = blockIdx.y * 64 + wave * 16;

  v8f acc = {};
#pragma unroll
  for (int k0 = 0; k0 < K; k0 += 32) {
    const int m  = row0 + (lane & 15);
    const int kb = k0 + ((lane >> 4) << 3);
    // A: two contiguous 8-float runs per lane -> four float4 loads
    const float* Ar = A + (size_t)m * K + kb;
    float4 a0 = ld4(Ar), a1 = ld4(Ar + 4), a2 = ld4(Ar + 16), a3 = ld4(Ar + 20);
    if (AOP == 1) {
      float4* as[4] = {&a0, &a1, &a2, &a3};
#pragma unroll
      for (int q = 0; q < 4; ++q) {
        float4& t = *as[q];
        t.x = fmaxf(0.25f * t.x, 0.f); t.y = fmaxf(0.25f * t.y, 0.f);
        t.z = fmaxf(0.25f * t.z, 0.f); t.w = fmaxf(0.25f * t.w, 0.f);
      }
    }
    if (HASAG) {
      const float* Gr = agate + (size_t)m * K + kb;
      float4 g0 = ld4(Gr), g1 = ld4(Gr + 4), g2 = ld4(Gr + 16), g3 = ld4(Gr + 20);
      a0.x *= g0.x; a0.y *= g0.y; a0.z *= g0.z; a0.w *= g0.w;
      a1.x *= g1.x; a1.y *= g1.y; a1.z *= g1.z; a1.w *= g1.w;
      a2.x *= g2.x; a2.y *= g2.y; a2.z *= g2.z; a2.w *= g2.w;
      a3.x *= g3.x; a3.y *= g3.y; a3.z *= g3.z; a3.w *= g3.w;
    }
    v16h a;
    a[0]=(_Float16)a0.x; a[1]=(_Float16)a0.y; a[2]=(_Float16)a0.z; a[3]=(_Float16)a0.w;
    a[4]=(_Float16)a1.x; a[5]=(_Float16)a1.y; a[6]=(_Float16)a1.z; a[7]=(_Float16)a1.w;
    a[8]=(_Float16)a2.x; a[9]=(_Float16)a2.y; a[10]=(_Float16)a2.z; a[11]=(_Float16)a2.w;
    a[12]=(_Float16)a3.x; a[13]=(_Float16)a3.y; a[14]=(_Float16)a3.z; a[15]=(_Float16)a3.w;

    // B: one base + 16 immediate-offset loads (coalesced across lanes)
    const float* Wb = W + (size_t)(k0 + ((lane >> 4) << 3)) * NOUT
                        + col0 + (lane & 15);
    v16h b;
#pragma unroll
    for (int e = 0; e < 16; ++e)
      b[e] = (_Float16)Wb[((e & 7) + ((e >> 3) << 4)) * NOUT];

    acc = __builtin_amdgcn_wmma_f32_16x16x32_f16(false, a, false, b,
                                                 (short)0, acc, false, false);
  }

  const int nc = col0 + (lane & 15);
#pragma unroll
  for (int r = 0; r < 8; ++r) {
    const int mr = row0 + r + ((lane >> 4) << 3);
    const size_t di = (size_t)mr * NOUT + nc;
    float x = acc[r];
    if (EPI == 0)      D[di] = x;
    else if (EPI == 1) D[di] = 1.0f / (1.0f + __expf(-x));
    else if (EPI == 2) D[di] += x * gate[(size_t)mr * NOUT + nc];
    else if (EPI == 3) D[di] = x * D[di];
    else               D[di] += x;
  }
}

// ---------------- LayerNorm over last dim D (32 lanes per row, wave32)
__global__ __launch_bounds__(128) void k_layernorm(
    const float* __restrict__ X, const float* __restrict__ g,
    const float* __restrict__ b, float* __restrict__ Y,
    int D, int M, float scale, int accum)
{
  int wid  = (int)((blockIdx.x * (size_t)blockDim.x + threadIdx.x) >> 5);
  int lane = threadIdx.x & 31;
  if (wid >= M) return;
  const float* x = X + (size_t)wid * D;
  int per = D >> 5;                       // 2 (D=64) or 4 (D=128)
  float s = 0.f, s2 = 0.f, vals[4];
#pragma unroll
  for (int i = 0; i < 4; ++i) {
    if (i < per) {
      float v = x[lane + (i << 5)];
      vals[i] = v; s += v; s2 += v * v;
    }
  }
#pragma unroll
  for (int off = 16; off >= 1; off >>= 1) {
    s  += __shfl_xor(s,  off, 32);
    s2 += __shfl_xor(s2, off, 32);
  }
  float inv = 1.0f / (float)D;
  float mu  = s * inv;
  float var = s2 * inv - mu * mu;
  float rs  = rsqrtf(var + EPSL);
  float* y = Y + (size_t)wid * D;
  for (int i = 0; i < per; ++i) {
    int c = lane + (i << 5);
    float o = (vals[i] - mu) * rs * g[c] + b[c];
    if (accum) y[c] += scale * o; else y[c] = o;
  }
}

// ---------------- trimul einsum: out[i,j,d] = sum_k a[.,.,d]*b[.,.,d]
// DIR 0 (outgoing): a[i,k]*b[j,k]   DIR 1 (incoming): a[k,i]*b[k,j]
// pg: NN x 256 f32 (a = [:,0:128], b = [:,128:256]); out: NN x 128 f32
template <int DIR>
__global__ __launch_bounds__(32) void k_trimul_einsum(
    const float* __restrict__ pg, float* __restrict__ out)
{
  const int lane = threadIdx.x & 31;
  const int i0 = blockIdx.x * 16, j0 = blockIdx.y * 16, d = blockIdx.z;
  // element stride along k (in floats): 256 (DIR 0) or 160*256 (DIR 1)
  const size_t KS = (DIR == 0) ? 256u : (size_t)NSEQ * 256u;
  v8f acc = {};
#pragma unroll
  for (int k0 = 0; k0 < NSEQ; k0 += 32) {
    const int mi = i0 + (lane & 15);
    const int kb = k0 + ((lane >> 4) << 3);
    const float* Pa = (DIR == 0) ? (pg + (size_t)(mi * NSEQ + kb) * 256 + d)
                                 : (pg + (size_t)(kb * NSEQ + mi) * 256 + d);
    v16h a;
#pragma unroll
    for (int e = 0; e < 16; ++e) {
      const size_t koff = (size_t)((e & 7) + ((e >> 3) << 4)) * KS;
      a[e] = (_Float16)Pa[koff];
    }
    const int nj  = j0 + (lane & 15);
    const int kb2 = k0 + ((lane >> 4) << 3);
    const float* Pb = (DIR == 0) ? (pg + (size_t)(nj * NSEQ + kb2) * 256 + 128 + d)
                                 : (pg + (size_t)(kb2 * NSEQ + nj) * 256 + 128 + d);
    v16h b;
#pragma unroll
    for (int e = 0; e < 16; ++e)
      b[e] = (_Float16)Pb[(size_t)((e & 7) + ((e >> 3) << 4)) * KS];

    acc = __builtin_amdgcn_wmma_f32_16x16x32_f16(false, a, false, b,
                                                 (short)0, acc, false, false);
  }
  const int nc = j0 + (lane & 15);
#pragma unroll
  for (int r = 0; r < 8; ++r) {
    const int mr = i0 + r + ((lane >> 4) << 3);
    out[(size_t)(mr * NSEQ + nc) * 128 + d] = acc[r];
  }
}

// ---------------- triangle attention for one (row n, head h)
// qkv: NN x 384 (q|k|v each 128 = 4 heads * 32), biasb: NN x 4,
// out: NN x 128 (h*32 + d). Full 160x160 scores kept in LDS (CDNA5 320KB/WGP).
__global__ __launch_bounds__(320) void k_attention(
    const float* __restrict__ qkv, const float* __restrict__ biasb,
    const float* __restrict__ tm, float* __restrict__ out)
{
  extern __shared__ char smem[];
  _Float16* q16 = (_Float16*)smem;              // 160*32
  _Float16* k16 = q16 + NSEQ * 32;              // 160*32
  _Float16* v16 = k16 + NSEQ * 32;              // 160*32
  float*    S   = (float*)(v16 + NSEQ * 32);    // 160*160 f32
  _Float16* P   = (_Float16*)(S + NSEQ * NSEQ); // 160*160 f16

  const int n = blockIdx.x, h = blockIdx.y;
  const int tid = threadIdx.x;
  for (int idx = tid; idx < NSEQ * 32; idx += 320) {
    int j = idx >> 5, d = idx & 31;
    size_t base = ((size_t)(n * NSEQ + j)) * 384 + h * 32 + d;
    q16[idx] = (_Float16)qkv[base];
    k16[idx] = (_Float16)qkv[base + 128];
    v16[idx] = (_Float16)qkv[base + 256];
  }
  __syncthreads();

  const int wave = tid >> 5, lane = tid & 31;
  // ---- scores: S = Q K^T / sqrt(32) + bias, masked
  {
    const int i0 = wave * 16;
    v16h aq;
    const int mi = i0 + (lane & 15);
    const int kb = (lane >> 4) << 3;
#pragma unroll
    for (int e = 0; e < 16; ++e) {
      int k = kb + (e & 7) + ((e >> 3) << 4);
      aq[e] = q16[mi * 32 + k];
    }
    const float tmn = tm[n];
#pragma unroll
    for (int jt = 0; jt < 10; ++jt) {
      const int j0 = jt * 16;
      v16h bk;
      const int nj  = j0 + (lane & 15);
      const int kb2 = (lane >> 4) << 3;
#pragma unroll
      for (int e = 0; e < 16; ++e)
        bk[e] = k16[nj * 32 + kb2 + (e & 7) + ((e >> 3) << 4)];
      v8f acc = {};
      acc = __builtin_amdgcn_wmma_f32_16x16x32_f16(false, aq, false, bk,
                                                   (short)0, acc, false, false);
      const int nc = j0 + (lane & 15);
      const float mj = tm[nc];
#pragma unroll
      for (int r = 0; r < 8; ++r) {
        int mr = i0 + r + ((lane >> 4) << 3);
        float s = acc[r] * 0.17677669529663687f      // 1/sqrt(32)
                + biasb[(size_t)(mr * NSEQ + nc) * 4 + h];
        if (tmn * mj <= 0.f) s = -1e30f;
        S[mr * NSEQ + nc] = s;
      }
    }
  }
  __syncthreads();
  // ---- row softmax -> f16 probabilities
  if (tid < NSEQ) {
    float m = -3.0e38f;
    for (int j = 0; j < NSEQ; ++j) m = fmaxf(m, S[tid * NSEQ + j]);
    float sum = 0.f;
    for (int j = 0; j < NSEQ; ++j) sum += __expf(S[tid * NSEQ + j] - m);
    float rinv = 1.0f / sum;
    for (int j = 0; j < NSEQ; ++j)
      P[tid * NSEQ + j] = (_Float16)(__expf(S[tid * NSEQ + j] - m) * rinv);
  }
  __syncthreads();
  // ---- out = P @ V  (K = 160, two 16-wide d tiles)
  {
    const int i0 = wave * 16;
    v8f acc0 = {}, acc1 = {};
#pragma unroll
    for (int k0 = 0; k0 < NSEQ; k0 += 32) {
      v16h ap;
      const int mi = i0 + (lane & 15);
      const int kb = k0 + ((lane >> 4) << 3);
#pragma unroll
      for (int e = 0; e < 16; ++e) {
        int k = kb + (e & 7) + ((e >> 3) << 4);
        ap[e] = P[mi * NSEQ + k];
      }
      const int kb2 = k0 + ((lane >> 4) << 3);
      v16h bv0, bv1;
#pragma unroll
      for (int e = 0; e < 16; ++e) {
        bv0[e] = v16[(kb2 + (e & 7) + ((e >> 3) << 4)) * 32 + (lane & 15)];
        bv1[e] = v16[(kb2 + (e & 7) + ((e >> 3) << 4)) * 32 + 16 + (lane & 15)];
      }
      acc0 = __builtin_amdgcn_wmma_f32_16x16x32_f16(false, ap, false, bv0,
                                                    (short)0, acc0, false, false);
      acc1 = __builtin_amdgcn_wmma_f32_16x16x32_f16(false, ap, false, bv1,
                                                    (short)0, acc1, false, false);
    }
#pragma unroll
    for (int r = 0; r < 8; ++r) {
      int mr = i0 + r + ((lane >> 4) << 3);
      size_t base = ((size_t)(n * NSEQ + mr)) * 128 + h * 32;
      out[base + (lane & 15)]      = acc0[r];
      out[base + 16 + (lane & 15)] = acc1[r];
    }
  }
}

// ---------------- small elementwise helpers
__global__ void k_caa(const float* __restrict__ Wa, int aa, float* __restrict__ caa) {
  int t = threadIdx.x;
  if (t < 64) caa[t] = Wa[(44 + aa) * 64 + t] + Wa[(76 + aa) * 64 + t];
}
__global__ void k_initv(const float* __restrict__ zproj, const float* __restrict__ caa,
                        float* __restrict__ v) {
  size_t idx = blockIdx.x * (size_t)blockDim.x + threadIdx.x;
  if (idx < (size_t)NN * 64) v[idx] = zproj[idx] + caa[idx & 63];
}
__global__ void k_transpose64(const float* __restrict__ src, float* __restrict__ dst) {
  size_t idx = blockIdx.x * (size_t)blockDim.x + threadIdx.x;
  if (idx >= (size_t)NN * 64) return;
  int p = (int)(idx >> 6), c = (int)(idx & 63);
  int i = p / NSEQ, j = p % NSEQ;
  dst[idx] = src[((size_t)(j * NSEQ + i) << 6) + c];
}
__global__ void k_silu(const float* __restrict__ big, float* __restrict__ ob) {
  size_t idx = blockIdx.x * (size_t)blockDim.x + threadIdx.x;
  if (idx >= (size_t)NN * 128) return;
  int p = (int)(idx >> 7), c = (int)(idx & 127);
  float a = big[(size_t)p * 256 + c];
  float b = big[(size_t)p * 256 + 128 + c];
  ob[idx] = (a / (1.0f + __expf(-a))) * b;
}
__global__ void k_bias(const float* __restrict__ h, const float* __restrict__ Wb,
                       float* __restrict__ biasb) {
  size_t idx = blockIdx.x * (size_t)blockDim.x + threadIdx.x;
  if (idx >= (size_t)NN * 4) return;
  int p = (int)(idx >> 2), hh = (int)(idx & 3);
  const float* hp = h + (size_t)p * 64;
  float s = 0.f;
#pragma unroll
  for (int k = 0; k < 64; ++k) s += hp[k] * Wb[k * 4 + hh];
  biasb[idx] = s;
}

// ---------------------------------------------------------------------------
extern "C" void kernel_launch(void* const* d_in, const int* in_sizes, int n_in,
                              void* d_out, int out_size, void* d_ws, size_t ws_size,
                              hipStream_t stream) {
  (void)in_sizes; (void)n_in; (void)out_size; (void)ws_size;
  const float* z  = (const float*)d_in[0];
  const float* tm = (const float*)d_in[2];   // token_mask (chain_indices unused: feat2d == 0)
  auto P = [&](int i) { return (const float*)d_in[i]; };
  // params flattened depth-first in setup_inputs() insertion order:
  const int I_ZN_G = 3, I_ZN_B = 4, I_WZ = 5, I_WA = 6;
  // block b at 7 + b*32:
  //   tmo +0..7  [ln_in_g, ln_in_b, Wp, Wg, ln_out_g, ln_out_b, Wo, Wgo]
  //   tmi +8..15
  //   tas +16..21 [ng, nb, Wqkv, Wb, Wout, Wgate]     tae +22..27
  //   tr_ng +28, tr_nb +29, W1 +30, W2 +31
  const int I_ON_G = 71, I_ON_B = 72, I_WOUT = 73;

  float* ws = (float*)d_ws;
  size_t off = 0;
  auto alloc = [&](size_t n) { float* p = ws + off; off += n; return p; };
  float* zproj = alloc((size_t)NN * 64);
  float* v     = alloc((size_t)NN * 64);
  float* vt    = alloc((size_t)NN * 64);
  float* hb    = alloc((size_t)NN * 64);    // LN(D=64) output
  float* h2    = alloc((size_t)NN * 128);   // LN(D=128) output
  float* gout  = alloc((size_t)NN * 64);    // sigmoid(h @ Wgo)
  float* u     = alloc((size_t)NN * 64);    // template accumulator
  float* big   = alloc((size_t)NN * 384);   // pg / qkv / W1-out
  float* ob    = alloc((size_t)NN * 128);   // einsum-out / attn-out / silu-out
  float* gateb = alloc((size_t)NN * 128);   // sigmoid(h @ Wgate)
  float* biasb = alloc((size_t)NN * 4);
  float* caa   = alloc(64);

  const dim3 blk128(128);
  auto LN = [&](const float* X, const float* g, const float* b, float* Y,
                int D, float sc, int acc) {
    k_layernorm<<<dim3(NN / 4), blk128, 0, stream>>>(X, g, b, Y, D, NN, sc, acc);
  };
  auto grid = [&](int Nout) { return dim3(NN / 16, Nout / 64); };

  // zproj = LN(z) @ Wz   (K=128 -> 64)
  LN(z, P(I_ZN_G), P(I_ZN_B), h2, 128, 1.f, 0);
  k_gemm<128, 64, 0, false, 0><<<grid(64), blk128, 0, stream>>>(
      h2, P(I_WZ), zproj, nullptr, nullptr);
  hipMemsetAsync(u, 0, (size_t)NN * 64 * sizeof(float), stream);

  // AATYPE_PER_SLOT = [31,0,0,0]: slots 1..3 identical -> 2 passes, weights {1,3}
  const int   aas[2] = {31, 0};
  const float uw[2]  = {1.f, 3.f};
  for (int pass = 0; pass < 2; ++pass) {
    k_caa<<<dim3(1), dim3(64), 0, stream>>>(P(I_WA), aas[pass], caa);
    k_initv<<<dim3(NN * 64 / 256), dim3(256), 0, stream>>>(zproj, caa, v);

    for (int b = 0; b < 2; ++b) {
      int B0 = 7 + b * 32;
      // ---- trimul outgoing (t=0) / incoming (t=1)
      for (int t = 0; t < 2; ++t) {
        int tb = B0 + t * 8;
        LN(v, P(tb + 0), P(tb + 1), hb, 64, 1.f, 0);
        k_gemm<64, 256, 1, false, 0><<<grid(256), blk128, 0, stream>>>(
            hb, P(tb + 3), big, nullptr, nullptr);          // sigmoid(h@Wg)
        k_gemm<64, 256, 3, false, 0><<<grid(256), blk128, 0, stream>>>(
            hb, P(tb + 2), big, nullptr, nullptr);          // *= h@Wp
        k_gemm<64, 64, 1, false, 0><<<grid(64), blk128, 0, stream>>>(
            hb, P(tb + 7), gout, nullptr, nullptr);         // sigmoid(h@Wgo)
        if (t == 0)
          k_trimul_einsum<0><<<dim3(10, 10, 128), dim3(32), 0, stream>>>(big, ob);
        else
          k_trimul_einsum<1><<<dim3(10, 10, 128), dim3(32), 0, stream>>>(big, ob);
        LN(ob, P(tb + 4), P(tb + 5), h2, 128, 1.f, 0);
        k_gemm<128, 64, 2, false, 0><<<grid(64), blk128, 0, stream>>>(
            h2, P(tb + 6), v, gout, nullptr);               // v += (LN@Wo)*gout
      }
      // ---- triatt starting (t=0 on v) / ending (t=1 on v^T)
      for (int t = 0; t < 2; ++t) {
        int ab = B0 + 16 + t * 6;
        float* cur = (t == 0) ? v : vt;
        if (t == 1)
          k_transpose64<<<dim3(NN * 64 / 256), dim3(256), 0, stream>>>(v, vt);
        LN(cur, P(ab + 0), P(ab + 1), hb, 64, 1.f, 0);
        k_gemm<64, 384, 0, false, 0><<<grid(384), blk128, 0, stream>>>(
            hb, P(ab + 2), big, nullptr, nullptr);          // qkv
        k_bias<<<dim3(NN * 4 / 256), dim3(256), 0, stream>>>(hb, P(ab + 3), biasb);
        k_gemm<64, 128, 1, false, 0><<<grid(128), blk128, 0, stream>>>(
            hb, P(ab + 5), gateb, nullptr, nullptr);        // sigmoid(h@Wgate)
        k_attention<<<dim3(NSEQ, 4), dim3(320),
                      3 * NSEQ * 32 * 2 + NSEQ * NSEQ * 4 + NSEQ * NSEQ * 2,
                      stream>>>(big, biasb, tm, ob);
        k_gemm<128, 64, 4, true, 0><<<grid(64), blk128, 0, stream>>>(
            ob, P(ab + 4), cur, nullptr, gateb);            // cur += (o*gate)@Wout
        if (t == 1)
          k_transpose64<<<dim3(NN * 64 / 256), dim3(256), 0, stream>>>(vt, v);
      }
      // ---- transition
      LN(v, P(B0 + 28), P(B0 + 29), hb, 64, 1.f, 0);
      k_gemm<64, 256, 0, false, 0><<<grid(256), blk128, 0, stream>>>(
          hb, P(B0 + 30), big, nullptr, nullptr);
      k_silu<<<dim3(NN * 128 / 256), dim3(256), 0, stream>>>(big, ob);
      k_gemm<128, 64, 4, false, 0><<<grid(64), blk128, 0, stream>>>(
          ob, P(B0 + 31), v, nullptr, nullptr);
    }
    LN(v, P(I_ON_G), P(I_ON_B), u, 64, uw[pass], 1);   // u += w * LN(v)
  }
  // out = relu(u/4) @ Wout   (aop=1 applies relu(0.25x) on the A operand)
  k_gemm<64, 128, 0, false, 1><<<grid(128), blk128, 0, stream>>>(
      u, P(I_WOUT), (float*)d_out, nullptr, nullptr);
}
